// MPASAGEChannel_45724221833421
// MI455X (gfx1250) — matrix operand order repaired and checked by reading, compiled, not run on hardware
//
#include <hip/hip_runtime.h>
#include <hip/hip_bf16.h>

typedef __attribute__((ext_vector_type(16))) _Float16 v16h;
typedef __attribute__((ext_vector_type(8)))  float    v8f;
typedef _Float16 half_t;

#define NN 50000
#define EE 1600000
#define FD 64

// ---------------------------------------------------------------------------
// Zero the accumulation regions of the workspace (agg1, agg2, deg0, deg1).
__global__ void zero_ws_kernel(float* __restrict__ p, int n) {
    int i = blockIdx.x * blockDim.x + threadIdx.x;
    if (i < n) p[i] = 0.0f;
}

// ---------------------------------------------------------------------------
// Convert the four 64x64 f32 weight matrices to f16, transposed to N-major
// (dst[n*64+k] = src[k*64+n]) so the WMMA B-matrix per-lane K-run is a
// contiguous 32-byte v16h load from LDS.
__global__ void prep_weights_kernel(const float* __restrict__ W1,
                                    const float* __restrict__ R1,
                                    const float* __restrict__ W2,
                                    const float* __restrict__ R2,
                                    half_t* __restrict__ dst) {
    int idx = blockIdx.x * blockDim.x + threadIdx.x;     // 4 * 4096
    if (idx >= 4 * 64 * 64) return;
    int w = idx >> 12;
    int r = idx & 4095;
    int n = r >> 6;
    int k = r & 63;
    const float* src = (w & 2) ? ((w & 1) ? R2 : W2) : ((w & 1) ? R1 : W1);
    dst[(w << 12) + n * 64 + k] = (half_t)src[k * 64 + n];
}

// ---------------------------------------------------------------------------
// Edge scatter-add: 16 lanes per edge, float4 gather of the source row,
// 4 scalar f32 atomics into the target accumulator (L2-resident). Lane 0
// also bumps the degree counter.
__global__ void scatter_mean_kernel(const float* __restrict__ feat,
                                    const long long* __restrict__ ei,
                                    float* __restrict__ agg,
                                    float* __restrict__ deg) {
    int idx = blockIdx.x * blockDim.x + threadIdx.x;
    int e = idx >> 4;
    int l = idx & 15;
    if (e >= EE) return;
    int src = (int)ei[e];          // edge_index[0][e]
    int dst = (int)ei[EE + e];     // edge_index[1][e]
    const float4 v = *(const float4*)(feat + (size_t)src * FD + l * 4);
    float* a = agg + (size_t)dst * FD + l * 4;
    atomicAdd(a + 0, v.x);
    atomicAdd(a + 1, v.y);
    atomicAdd(a + 2, v.z);
    atomicAdd(a + 3, v.w);
    if (l == 0) atomicAdd(deg + dst, 1.0f);
}

// ---------------------------------------------------------------------------
// Fused SAGE layer GEMM:
//   out = epilogue( (agg/deg) @ W  +  xin @ R  + bias )
// One wave32 owns 16 node rows; 8 waves / block -> 128 rows per block.
// 16x16x32 f16 WMMA, K=64 as two K-steps, agg- and x-branches chained into
// one f32 accumulator. RELU for layer 1; NORMALIZE (row-L2) for layer 2.
template <bool RELU, bool NORMALIZE>
__global__ void sage_gemm_kernel(const float*  __restrict__ agg,
                                 const float*  __restrict__ deg,
                                 const float*  __restrict__ xin,
                                 const half_t* __restrict__ Wt,   // [n][k] f16
                                 const half_t* __restrict__ Rt,   // [n][k] f16
                                 const float*  __restrict__ bias,
                                 float*        __restrict__ out) {
    __shared__ __attribute__((aligned(32))) half_t sW[64 * 64];
    __shared__ __attribute__((aligned(32))) half_t sR[64 * 64];
    for (int i = threadIdx.x; i < 64 * 64; i += blockDim.x) {
        sW[i] = Wt[i];
        sR[i] = Rt[i];
    }
    __syncthreads();

    const int lane = threadIdx.x & 31;
    const int wave = threadIdx.x >> 5;
    const int m    = lane & 15;     // A row within tile / B,C column
    const int hi   = lane >> 4;     // lane half
    const int row0 = blockIdx.x * 128 + wave * 16;

    // Clamp tail rows so EXEC stays all-1s through the WMMAs.
    int node   = row0 + m;
    int node_c = node < NN ? node : NN - 1;

    const float inv_deg = 1.0f / fmaxf(deg[node_c], 1.0f);

    // A-matrix (16x32 f16): per lane, element i holds
    // K = kt*32 + hi*16 + i of row m -> two contiguous
    // 8-float runs per K-tile.
    v16h a_agg[2], a_x[2];
#pragma unroll
    for (int kt = 0; kt < 2; ++kt) {
#pragma unroll
        for (int r = 0; r < 2; ++r) {
            const int kb = kt * 32 + hi * 16 + r * 8;
            const float4 f0 = *(const float4*)(agg + (size_t)node_c * FD + kb);
            const float4 f1 = *(const float4*)(agg + (size_t)node_c * FD + kb + 4);
            const float4 g0 = *(const float4*)(xin + (size_t)node_c * FD + kb);
            const float4 g1 = *(const float4*)(xin + (size_t)node_c * FD + kb + 4);
            const int b = r * 8;
            a_agg[kt][b + 0] = (half_t)(f0.x * inv_deg);
            a_agg[kt][b + 1] = (half_t)(f0.y * inv_deg);
            a_agg[kt][b + 2] = (half_t)(f0.z * inv_deg);
            a_agg[kt][b + 3] = (half_t)(f0.w * inv_deg);
            a_agg[kt][b + 4] = (half_t)(f1.x * inv_deg);
            a_agg[kt][b + 5] = (half_t)(f1.y * inv_deg);
            a_agg[kt][b + 6] = (half_t)(f1.z * inv_deg);
            a_agg[kt][b + 7] = (half_t)(f1.w * inv_deg);
            a_x[kt][b + 0] = (half_t)g0.x;
            a_x[kt][b + 1] = (half_t)g0.y;
            a_x[kt][b + 2] = (half_t)g0.z;
            a_x[kt][b + 3] = (half_t)g0.w;
            a_x[kt][b + 4] = (half_t)g1.x;
            a_x[kt][b + 5] = (half_t)g1.y;
            a_x[kt][b + 6] = (half_t)g1.z;
            a_x[kt][b + 7] = (half_t)g1.w;
        }
    }

    // Four 16-wide N-tiles; B element i = Bmat[K = kt*32 + hi*16 + i][N = m],
    // i.e. a contiguous 32B v16h run in the transposed LDS copy.
    v8f acc[4];
#pragma unroll
    for (int nt = 0; nt < 4; ++nt) {
        const int nrow = (nt * 16 + m) * 64;
        const v16h bW0 = *(const v16h*)(sW + nrow + hi * 16);
        const v16h bW1 = *(const v16h*)(sW + nrow + 32 + hi * 16);
        const v16h bR0 = *(const v16h*)(sR + nrow + hi * 16);
        const v16h bR1 = *(const v16h*)(sR + nrow + 32 + hi * 16);
        v8f c = {};
        c = __builtin_amdgcn_wmma_f32_16x16x32_f16(false, a_agg[0], false, bW0,
                                                   (short)0, c, false, false);
        c = __builtin_amdgcn_wmma_f32_16x16x32_f16(false, a_agg[1], false, bW1,
                                                   (short)0, c, false, false);
        c = __builtin_amdgcn_wmma_f32_16x16x32_f16(false, a_x[0], false, bR0,
                                                   (short)0, c, false, false);
        c = __builtin_amdgcn_wmma_f32_16x16x32_f16(false, a_x[1], false, bR1,
                                                   (short)0, c, false, false);
        const float bv = bias[nt * 16 + m];
#pragma unroll
        for (int j = 0; j < 8; ++j) {
            float v = c[j] + bv;
            if (RELU) v = fmaxf(v, 0.0f);
            c[j] = v;
        }
        acc[nt] = c;
    }

    // Row-wise L2 normalize fused into the epilogue. C layout: element j of
    // lane L holds row M = j + 8*hi, col N = m. Reduce sum-of-squares over
    // the 16 lanes of each half with a butterfly.
    float scale[8];
    if (NORMALIZE) {
        float p[8];
#pragma unroll
        for (int j = 0; j < 8; ++j)
            p[j] = acc[0][j] * acc[0][j] + acc[1][j] * acc[1][j] +
                   acc[2][j] * acc[2][j] + acc[3][j] * acc[3][j];
#pragma unroll
        for (int off = 1; off <= 8; off <<= 1) {
#pragma unroll
            for (int j = 0; j < 8; ++j) p[j] += __shfl_xor(p[j], off, 32);
        }
#pragma unroll
        for (int j = 0; j < 8; ++j)
            scale[j] = 1.0f / fmaxf(sqrtf(p[j]), 1e-12f);
    }

#pragma unroll
    for (int j = 0; j < 8; ++j) {
        const int orow = row0 + j + 8 * hi;
        if (orow < NN) {
#pragma unroll
            for (int nt = 0; nt < 4; ++nt) {
                float v = acc[nt][j];
                if (NORMALIZE) v *= scale[j];
                out[(size_t)orow * FD + nt * 16 + m] = v;
            }
        }
    }
}

// ---------------------------------------------------------------------------
extern "C" void kernel_launch(void* const* d_in, const int* in_sizes, int n_in,
                              void* d_out, int out_size, void* d_ws, size_t ws_size,
                              hipStream_t stream) {
    const float*     x   = (const float*)d_in[0];
    const long long* ei0 = (const long long*)d_in[1];
    const long long* ei1 = (const long long*)d_in[2];
    const float*     W1  = (const float*)d_in[3];
    const float*     b1  = (const float*)d_in[4];
    const float*     R1  = (const float*)d_in[5];
    const float*     W2  = (const float*)d_in[6];
    const float*     b2  = (const float*)d_in[7];
    const float*     R2  = (const float*)d_in[8];
    float*           out = (float*)d_out;

    float* ws   = (float*)d_ws;
    float* agg1 = ws;                       // N*64
    float* agg2 = agg1 + (size_t)NN * FD;   // N*64
    float* deg0 = agg2 + (size_t)NN * FD;   // N
    float* deg1 = deg0 + NN;                // N
    float* h    = deg1 + NN;                // N*64
    half_t* wts = (half_t*)(h + (size_t)NN * FD);  // 4*4096 f16
    half_t* Wt1 = wts;
    half_t* Rt1 = wts + 4096;
    half_t* Wt2 = wts + 8192;
    half_t* Rt2 = wts + 12288;

    // agg1, agg2, deg0, deg1 are contiguous -> one zeroing pass.
    const int zn = 2 * NN * FD + 2 * NN;
    zero_ws_kernel<<<(zn + 255) / 256, 256, 0, stream>>>(ws, zn);
    prep_weights_kernel<<<(4 * 64 * 64 + 255) / 256, 256, 0, stream>>>(
        W1, R1, W2, R2, wts);

    const int sthreads = EE * 16;
    const int sgrid = (sthreads + 255) / 256;
    const int ggrid = (NN + 127) / 128;

    // Layer 1: mean-aggregate x over edge_index_0, then relu(agg@W1+b1+x@R1).
    scatter_mean_kernel<<<sgrid, 256, 0, stream>>>(x, ei0, agg1, deg0);
    sage_gemm_kernel<true, false><<<ggrid, 256, 0, stream>>>(
        agg1, deg0, x, Wt1, Rt1, b1, h);

    // Layer 2: mean-aggregate h over edge_index_1, then agg@W2+b2+h@R2,
    // fused row-wise L2 normalize into the epilogue.
    scatter_mean_kernel<<<sgrid, 256, 0, stream>>>(h, ei1, agg2, deg1);
    sage_gemm_kernel<false, true><<<ggrid, 256, 0, stream>>>(
        agg2, deg1, h, Wt2, Rt2, b2, out);
}
